// TransformerBlock_13469017440329
// MI455X (gfx1250) — compile-verified, matrix operand order checked by reading, not run on hardware
//
#include <hip/hip_runtime.h>
#include <hip/hip_bf16.h>
#include <math.h>

// ---------------------------------------------------------------------------
// Types for CDNA5 WMMA (wave32): A/B = 16 x f16 per lane, C/D = 8 x f32.
// ---------------------------------------------------------------------------
typedef _Float16 v16h __attribute__((ext_vector_type(16)));
typedef _Float16 h8   __attribute__((ext_vector_type(8)));
typedef _Float16 h4   __attribute__((ext_vector_type(4)));
typedef float    v8f  __attribute__((ext_vector_type(8)));

// Problem constants (match reference)
#define B_   2
#define S_   2048
#define D_   1024
#define H_   16
#define DK_  64
#define DFF_ 4096
#define M_   (B_ * S_)   // 4096 rows

// ---------------------------------------------------------------------------
// Fragment load following the documented 16-bit A-matrix 16x32 layout:
//   lanes 0-15 : row M=(lane&15), K = kbase+{0..7} and kbase+{16..23}
//   lanes 16-31: row M=(lane&15), K = kbase+{8..15} and kbase+{24..31}
// The B operand of an NT GEMM (W stored N x K row-major) uses the same
// pattern with "row" = output column, so one helper serves both.
// ---------------------------------------------------------------------------
__device__ __forceinline__ v16h load_frag(const _Float16* base, int row, int ldk,
                                          int kbase, int lane) {
    const _Float16* p = base + (size_t)row * ldk + kbase + ((lane & 16) ? 8 : 0);
    h8 lo = *(const h8*)(p);
    h8 hi = *(const h8*)(p + 16);
    return __builtin_shufflevector(lo, hi, 0, 1, 2, 3, 4, 5, 6, 7,
                                   8, 9, 10, 11, 12, 13, 14, 15);
}

__device__ __forceinline__ v8f wmma_f16(v16h a, v16h b, v8f c) {
    // v_wmma_f32_16x16x32_f16
    return __builtin_amdgcn_wmma_f32_16x16x32_f16(false, a, false, b,
                                                  (short)0, c, false, false);
}

__device__ __forceinline__ float gelu_erf(float v) {
    return 0.5f * v * (1.0f + erff(v * 0.70710678118654752f));
}

// ---------------------------------------------------------------------------
// f32 -> f16 cast (weights, once)
// ---------------------------------------------------------------------------
__global__ void cast_f32_f16_kernel(const float* __restrict__ in,
                                    _Float16* __restrict__ out, size_t n) {
    size_t i = ((size_t)blockIdx.x * blockDim.x + threadIdx.x) * 4;
    if (i + 3 < n) {
        float4 v = *(const float4*)(in + i);
        h4 o = { (_Float16)v.x, (_Float16)v.y, (_Float16)v.z, (_Float16)v.w };
        *(h4*)(out + i) = o;
    }
}

// ---------------------------------------------------------------------------
// LayerNorm over D=1024, one block (256 thr = 8 waves) per row, f16 output.
// D = 256 threads * 4 -> one float4 per thread, fully vectorized.
// ---------------------------------------------------------------------------
__global__ void __launch_bounds__(256)
ln_cast_kernel(const float* __restrict__ x, const float* __restrict__ g,
               const float* __restrict__ b, _Float16* __restrict__ out) {
    const int row = blockIdx.x;
    const int i   = threadIdx.x * 4;
    const float4 v  = *(const float4*)(x + (size_t)row * D_ + i);
    const float4 gv = *(const float4*)(g + i);
    const float4 bv = *(const float4*)(b + i);

    float s  = v.x + v.y + v.z + v.w;
    float s2 = v.x * v.x + v.y * v.y + v.z * v.z + v.w * v.w;
    for (int m = 16; m >= 1; m >>= 1) {
        s  += __shfl_xor(s, m, 32);
        s2 += __shfl_xor(s2, m, 32);
    }
    __shared__ float ss[8], ss2[8];
    if ((threadIdx.x & 31) == 0) {
        ss[threadIdx.x >> 5]  = s;
        ss2[threadIdx.x >> 5] = s2;
    }
    __syncthreads();
    s = 0.f; s2 = 0.f;
#pragma unroll
    for (int w = 0; w < 8; w++) { s += ss[w]; s2 += ss2[w]; }
    const float mu   = s * (1.f / D_);
    const float var  = s2 * (1.f / D_) - mu * mu;
    const float rinv = rsqrtf(var + 1e-5f);

    h4 o = { (_Float16)((v.x - mu) * rinv * gv.x + bv.x),
             (_Float16)((v.y - mu) * rinv * gv.y + bv.y),
             (_Float16)((v.z - mu) * rinv * gv.z + bv.z),
             (_Float16)((v.w - mu) * rinv * gv.w + bv.w) };
    *(h4*)(out + (size_t)row * D_ + i) = o;
}

// ---------------------------------------------------------------------------
// WMMA GEMM: C[M,N] = epilogue(A[M,K] * W[N,K]^T + bias)
//   MODE 1: f16 out, split heads  -> q/k/v layout [B,H,S,DK]
//   MODE 2: f16 out, exact GELU   -> FFN hidden
//   MODE 3: f32 out, + residual   -> out-proj / FFN2
// Block tile 128x128, BK=64, 256 threads = 8 waves (2 x 4), wave tile 64x32.
// Double-buffered LDS; next tile staged in registers while WMMAs run, so
// global-load latency hides behind matrix math; one barrier per K-step.
// ---------------------------------------------------------------------------
template <int MODE>
__global__ void __launch_bounds__(256)
gemm_wmma_kernel(const _Float16* __restrict__ A, const _Float16* __restrict__ W,
                 const float* __restrict__ bias, const float* __restrict__ resid,
                 void* __restrict__ out, int M, int N, int K) {
    __shared__ __align__(16) _Float16 As[2][128 * 64];   // 16 KB each
    __shared__ __align__(16) _Float16 Bs[2][128 * 64];

    const int tid  = threadIdx.x;
    const int lane = tid & 31;
    const int wid  = tid >> 5;
    const int wm   = wid & 1;   // 0..1 : 64-row slice
    const int wn   = wid >> 1;  // 0..3 : 32-col slice
    const int m0   = blockIdx.y * 128;
    const int n0   = blockIdx.x * 128;

    // stage-load mapping: 128 rows x 64 cols, 8 f16 per load, 4 rows/thread
    const int lr = tid >> 3;         // 0..31
    const int lc = (tid & 7) << 3;   // 0..56

    v8f acc[4][2];
#pragma unroll
    for (int i = 0; i < 4; i++)
#pragma unroll
        for (int j = 0; j < 2; j++) acc[i][j] = (v8f)(0.f);

    uint4 aReg[4], bReg[4];
    // prologue: tile 0 -> regs -> LDS buf 0
#pragma unroll
    for (int p = 0; p < 4; p++) {
        aReg[p] = *(const uint4*)&A[(size_t)(m0 + lr + p * 32) * K + lc];
        bReg[p] = *(const uint4*)&W[(size_t)(n0 + lr + p * 32) * K + lc];
    }
#pragma unroll
    for (int p = 0; p < 4; p++) {
        *(uint4*)&As[0][(size_t)(lr + p * 32) * 64 + lc] = aReg[p];
        *(uint4*)&Bs[0][(size_t)(lr + p * 32) * 64 + lc] = bReg[p];
    }
    __syncthreads();

    int cur = 0;
    for (int kt = 0; kt < K; kt += 64) {
        const bool hasNext = (kt + 64 < K);
        if (hasNext) {   // issue next tile's global loads first (latency hide)
#pragma unroll
            for (int p = 0; p < 4; p++) {
                aReg[p] = *(const uint4*)&A[(size_t)(m0 + lr + p * 32) * K + kt + 64 + lc];
                bReg[p] = *(const uint4*)&W[(size_t)(n0 + lr + p * 32) * K + kt + 64 + lc];
            }
        }
        // compute on current buffer: 2 k-subtiles of 32, 8 WMMAs each
        const _Float16* Ac = As[cur];
        const _Float16* Bc = Bs[cur];
#pragma unroll
        for (int kk = 0; kk < 64; kk += 32) {
            v16h af[4], bf[2];
#pragma unroll
            for (int i = 0; i < 4; i++)
                af[i] = load_frag(Ac, wm * 64 + i * 16 + (lane & 15), 64, kk, lane);
#pragma unroll
            for (int j = 0; j < 2; j++)
                bf[j] = load_frag(Bc, wn * 32 + j * 16 + (lane & 15), 64, kk, lane);
#pragma unroll
            for (int i = 0; i < 4; i++)
#pragma unroll
                for (int j = 0; j < 2; j++)
                    acc[i][j] = wmma_f16(af[i], bf[j], acc[i][j]);
        }
        if (hasNext) {
            const int nxt = cur ^ 1;
#pragma unroll
            for (int p = 0; p < 4; p++) {
                *(uint4*)&As[nxt][(size_t)(lr + p * 32) * 64 + lc] = aReg[p];
                *(uint4*)&Bs[nxt][(size_t)(lr + p * 32) * 64 + lc] = bReg[p];
            }
        }
        __syncthreads();
        cur ^= 1;
    }

    // Epilogue. C layout: vgpr e -> M = e + (lane>=16 ? 8 : 0), N = lane&15.
    const int colL = lane & 15;
    const int mOff = (lane & 16) ? 8 : 0;
#pragma unroll
    for (int i = 0; i < 4; i++) {
#pragma unroll
        for (int j = 0; j < 2; j++) {
#pragma unroll
            for (int e = 0; e < 8; e++) {
                const int row = m0 + wm * 64 + i * 16 + e + mOff;
                const int col = n0 + wn * 32 + j * 16 + colL;
                float v = acc[i][j][e] + bias[col];
                if (MODE == 2) v = gelu_erf(v);
                if (MODE == 3) {
                    ((float*)out)[(size_t)row * N + col] =
                        resid[(size_t)row * N + col] + v;
                } else if (MODE == 1) {
                    // compile-time S_/H_ -> shifts, no integer division
                    const int bb = row >> 11;          // row / S_
                    const int ssd = row & (S_ - 1);
                    const int hh = col >> 6, dd = col & 63;
                    ((_Float16*)out)[(((size_t)(bb * H_ + hh) * S_ + ssd) << 6) + dd] =
                        (_Float16)v;
                } else {
                    ((_Float16*)out)[(size_t)row * N + col] = (_Float16)v;
                }
            }
        }
    }
}

// ---------------------------------------------------------------------------
// Flash-attention: grid (S/64, B*H), 128 threads = 4 waves.
// Each wave owns a 16-row Q tile; KV streamed in 64-row chunks through LDS.
// ---------------------------------------------------------------------------
__global__ void __launch_bounds__(128)
attn_kernel(const _Float16* __restrict__ Q, const _Float16* __restrict__ K,
            const _Float16* __restrict__ V, const unsigned char* __restrict__ mask,
            _Float16* __restrict__ ctx) {
    const int bh   = blockIdx.y;
    const int b    = bh >> 4;            // / H_
    const int h    = bh & (H_ - 1);
    const int lane = threadIdx.x & 31;
    const int wid  = threadIdx.x >> 5;          // 0..3
    const int q0   = blockIdx.x * 64 + wid * 16;
    const int mOff = (lane & 16) ? 8 : 0;
    const int colL = lane & 15;

    __shared__ __align__(16) _Float16 Ks[64 * 64];      // [kv][dk]
    __shared__ __align__(16) _Float16 Vt[64 * 64];      // [dk][kv] (transposed)
    __shared__ __align__(16) _Float16 Ps[4][16 * 64];   // per-wave P tile

    const _Float16* Qb = Q + (size_t)bh * S_ * DK_;
    const size_t kvb   = (size_t)bh * S_ * DK_;

    const v16h qf0 = load_frag(Qb, q0 + colL, DK_, 0, lane);
    const v16h qf1 = load_frag(Qb, q0 + colL, DK_, 32, lane);

    v8f o[4];
#pragma unroll
    for (int j = 0; j < 4; j++) o[j] = (v8f)(0.f);
    float mrow[8], lrow[8];
#pragma unroll
    for (int e = 0; e < 8; e++) { mrow[e] = -1e30f; lrow[e] = 0.f; }

    for (int c = 0; c < S_; c += 64) {
        __syncthreads();   // protect Ks/Vt from previous iteration's readers
        {
            const int rr0 = threadIdx.x >> 3;        // 0..15
            const int cc  = (threadIdx.x & 7) << 3;  // 0..56
#pragma unroll
            for (int p = 0; p < 4; p++) {
                const int rr = rr0 + p * 16;
                *(uint4*)&Ks[(size_t)rr * 64 + cc] =
                    *(const uint4*)&K[kvb + (size_t)(c + rr) * DK_ + cc];
                h8 vv = *(const h8*)&V[kvb + (size_t)(c + rr) * DK_ + cc];
#pragma unroll
                for (int e2 = 0; e2 < 8; e2++)
                    Vt[(size_t)(cc + e2) * 64 + rr] = vv[e2];
            }
        }
        __syncthreads();

        // scores: 16 x 64 = 4 WMMA column tiles, K-dim = DK (2 steps of 32)
        v8f sacc[4];
#pragma unroll
        for (int j = 0; j < 4; j++) {
            sacc[j] = (v8f)(0.f);
            sacc[j] = wmma_f16(qf0, load_frag(Ks, j * 16 + colL, 64, 0, lane), sacc[j]);
            sacc[j] = wmma_f16(qf1, load_frag(Ks, j * 16 + colL, 64, 32, lane), sacc[j]);
        }

        // scale + mask
#pragma unroll
        for (int j = 0; j < 4; j++) {
#pragma unroll
            for (int e = 0; e < 8; e++) {
                float sv = sacc[j][e] * 0.125f;  // 1/sqrt(64)
                const int qr = q0 + e + mOff;
                const int kv = c + j * 16 + colL;
                if (mask[((size_t)b * S_ + qr) * S_ + kv]) sv = -10000.f;
                sacc[j][e] = sv;
            }
        }

        // online softmax: rows live in 16-lane halves at fixed vgpr index e
#pragma unroll
        for (int e = 0; e < 8; e++) {
            float mx = fmaxf(fmaxf(sacc[0][e], sacc[1][e]),
                             fmaxf(sacc[2][e], sacc[3][e]));
#pragma unroll
            for (int xm = 1; xm < 16; xm <<= 1)
                mx = fmaxf(mx, __shfl_xor(mx, xm, 32));
            const float mnew  = fmaxf(mrow[e], mx);
            const float alpha = __expf(mrow[e] - mnew);
            float rs = 0.f;
#pragma unroll
            for (int j = 0; j < 4; j++) {
                float p = __expf(sacc[j][e] - mnew);
                sacc[j][e] = p;
                rs += p;
            }
#pragma unroll
            for (int xm = 1; xm < 16; xm <<= 1)
                rs += __shfl_xor(rs, xm, 32);
            lrow[e] = lrow[e] * alpha + rs;
            mrow[e] = mnew;
#pragma unroll
            for (int j = 0; j < 4; j++) o[j][e] *= alpha;
        }

        // re-fragment P through per-wave LDS (C-layout -> A-layout)
        _Float16* Pw = Ps[wid];
#pragma unroll
        for (int j = 0; j < 4; j++)
#pragma unroll
            for (int e = 0; e < 8; e++)
                Pw[(size_t)(e + mOff) * 64 + j * 16 + colL] = (_Float16)sacc[j][e];

        const v16h pf0 = load_frag(Pw, colL, 64, 0, lane);
        const v16h pf1 = load_frag(Pw, colL, 64, 32, lane);
#pragma unroll
        for (int j = 0; j < 4; j++) {
            o[j] = wmma_f16(pf0, load_frag(Vt, j * 16 + colL, 64, 0, lane), o[j]);
            o[j] = wmma_f16(pf1, load_frag(Vt, j * 16 + colL, 64, 32, lane), o[j]);
        }
    }

    // normalize + merge heads: ctx[b,s, h*64 + dk] (f16, feeds out-proj GEMM)
#pragma unroll
    for (int j = 0; j < 4; j++) {
#pragma unroll
        for (int e = 0; e < 8; e++) {
            const int qr = q0 + e + mOff;
            const int dk = j * 16 + colL;
            const float val = o[j][e] / lrow[e];
            ctx[((size_t)(b * S_ + qr)) * D_ + h * DK_ + dk] = (_Float16)val;
        }
    }
}

// ---------------------------------------------------------------------------
// Host-side orchestration
// ---------------------------------------------------------------------------
extern "C" void kernel_launch(void* const* d_in, const int* in_sizes, int n_in,
                              void* d_out, int out_size, void* d_ws, size_t ws_size,
                              hipStream_t stream) {
    // inputs (setup_inputs order)
    const float* x      = (const float*)d_in[0];
    const unsigned char* mask = (const unsigned char*)d_in[1];
    const float* Wq = (const float*)d_in[2];  const float* bq = (const float*)d_in[3];
    const float* Wk = (const float*)d_in[4];  const float* bk = (const float*)d_in[5];
    const float* Wv = (const float*)d_in[6];  const float* bv = (const float*)d_in[7];
    const float* Wo = (const float*)d_in[8];  const float* bo = (const float*)d_in[9];
    const float* W1 = (const float*)d_in[10]; const float* b1 = (const float*)d_in[11];
    const float* W2 = (const float*)d_in[12]; const float* b2 = (const float*)d_in[13];
    const float* g_attn = (const float*)d_in[14]; const float* b_attn = (const float*)d_in[15];
    const float* g_ffn  = (const float*)d_in[16]; const float* b_ffn  = (const float*)d_in[17];
    float* out = (float*)d_out;

    // workspace carve-out
    char* wsp = (char*)d_ws;
    auto take = [&](size_t bytes) { char* p = wsp; wsp += bytes; return p; };
    _Float16* wq16 = (_Float16*)take((size_t)D_ * D_ * 2);
    _Float16* wk16 = (_Float16*)take((size_t)D_ * D_ * 2);
    _Float16* wv16 = (_Float16*)take((size_t)D_ * D_ * 2);
    _Float16* wo16 = (_Float16*)take((size_t)D_ * D_ * 2);
    _Float16* w116 = (_Float16*)take((size_t)DFF_ * D_ * 2);
    _Float16* w216 = (_Float16*)take((size_t)D_ * DFF_ * 2);
    _Float16* nx16 = (_Float16*)take((size_t)M_ * D_ * 2);
    _Float16* q16  = (_Float16*)take((size_t)M_ * D_ * 2);
    _Float16* k16  = (_Float16*)take((size_t)M_ * D_ * 2);
    _Float16* v16  = (_Float16*)take((size_t)M_ * D_ * 2);
    _Float16* c16  = (_Float16*)take((size_t)M_ * D_ * 2);
    float*    x1   = (float*)take((size_t)M_ * D_ * 4);
    _Float16* nx2  = (_Float16*)take((size_t)M_ * D_ * 2);
    _Float16* h16  = (_Float16*)take((size_t)M_ * DFF_ * 2);

    // 1) cast weights to f16 (once; all GEMMs reuse)
    auto cast = [&](const float* src, _Float16* dst, size_t n) {
        cast_f32_f16_kernel<<<(unsigned)((n / 4 + 255) / 256), 256, 0, stream>>>(src, dst, n);
    };
    cast(Wq, wq16, (size_t)D_ * D_);
    cast(Wk, wk16, (size_t)D_ * D_);
    cast(Wv, wv16, (size_t)D_ * D_);
    cast(Wo, wo16, (size_t)D_ * D_);
    cast(W1, w116, (size_t)DFF_ * D_);
    cast(W2, w216, (size_t)D_ * DFF_);

    // 2) LN1 -> nx (f16)
    ln_cast_kernel<<<M_, 256, 0, stream>>>(x, g_attn, b_attn, nx16);

    // 3) Q/K/V projections (split-head epilogue -> [B,H,S,DK] f16)
    dim3 gProj(D_ / 128, M_ / 128);
    gemm_wmma_kernel<1><<<gProj, 256, 0, stream>>>(nx16, wq16, bq, nullptr, q16,
                                                   M_, D_, D_);
    gemm_wmma_kernel<1><<<gProj, 256, 0, stream>>>(nx16, wk16, bk, nullptr, k16,
                                                   M_, D_, D_);
    gemm_wmma_kernel<1><<<gProj, 256, 0, stream>>>(nx16, wv16, bv, nullptr, v16,
                                                   M_, D_, D_);

    // 4) fused flash attention -> ctx (f16, heads merged)
    dim3 gAttn(S_ / 64, B_ * H_);
    attn_kernel<<<gAttn, 128, 0, stream>>>(q16, k16, v16, mask, c16);

    // 5) out-proj + residual -> x1 (f32)
    gemm_wmma_kernel<3><<<gProj, 256, 0, stream>>>(c16, wo16, bo, x, x1,
                                                   M_, D_, D_);

    // 6) LN2 -> nx2 (f16)
    ln_cast_kernel<<<M_, 256, 0, stream>>>(x1, g_ffn, b_ffn, nx2);

    // 7) FFN1 + exact GELU -> h (f16)
    dim3 gF1(DFF_ / 128, M_ / 128);
    gemm_wmma_kernel<2><<<gF1, 256, 0, stream>>>(nx2, w116, b1, nullptr, h16,
                                                 M_, DFF_, D_);

    // 8) FFN2 + residual -> final output (f32)
    gemm_wmma_kernel<3><<<gProj, 256, 0, stream>>>(h16, w216, b2, x1, out,
                                                   M_, D_, DFF_);
}
